// Attention_6614249636281
// MI455X (gfx1250) — matrix operand order repaired and checked by reading, not compiled, not run on hardware
//
#include <hip/hip_runtime.h>
#include <hip/hip_bf16.h>

typedef __attribute__((ext_vector_type(16))) _Float16 v16h;
typedef __attribute__((ext_vector_type(8)))  _Float16 v8h;
typedef __attribute__((ext_vector_type(8)))  float    v8f;
typedef __attribute__((ext_vector_type(8)))  int      v8i;

#define BATCH 4
#define SEQ   4096
#define DIN   768
#define DHEAD 64

union V16U { v16h v; v8h h[2]; v8i i; };

// A-operand (16-bit, 16x32): per ISA table, lane half 'hi' holds two
// contiguous 8-half K-runs at offsets hi*8 and hi*8+16. Caller passes both.
static __device__ __forceinline__ v16h ldpair(const _Float16* p, int o0, int o1) {
    V16U u;
    u.h[0] = *(const v8h*)(p + o0);
    u.h[1] = *(const v8h*)(p + o1);
    return u.v;
}

// 16-bit operand (A or B) in this code's contiguous k order, not the ISA table order:
// lane half 'hi' holds one contiguous 16-half K-run (K = hi*16 + 0..15).
static __device__ __forceinline__ v16h ld16(const _Float16* p, int o) {
    V16U u;
    u.h[0] = *(const v8h*)(p + o);
    u.h[1] = *(const v8h*)(p + o + 8);
    return u.v;
}

static __device__ __forceinline__ v8f wmma16(v16h a, v16h b, v8f c) {
    return __builtin_amdgcn_wmma_f32_16x16x32_f16(
        /*neg_a=*/false, a, /*neg_b=*/false, b,
        /*c_mod=*/(short)0, c, /*reuse_a=*/false, /*reuse_b=*/false);
}

// ---------------------------------------------------------------------------
// Phase 0: convert + transpose weights to f16, Wt[o][i] = W[i][o].
// Fold the softmax scale 1/sqrt(64)=0.125 into Wq.
// ---------------------------------------------------------------------------
__global__ __launch_bounds__(256) void prep_w(const float* __restrict__ Wq,
                                              const float* __restrict__ Wk,
                                              const float* __restrict__ Wv,
                                              _Float16* __restrict__ Wtq,
                                              _Float16* __restrict__ Wtk,
                                              _Float16* __restrict__ Wtv) {
    int idx = blockIdx.x * blockDim.x + threadIdx.x;
    if (idx >= 3 * DIN * DHEAD) return;
    int m   = idx / (DIN * DHEAD);
    int rem = idx - m * (DIN * DHEAD);
    int i = rem / DHEAD;
    int o = rem - i * DHEAD;
    const float* W = (m == 0) ? Wq : ((m == 1) ? Wk : Wv);
    _Float16*   Wt = (m == 0) ? Wtq : ((m == 1) ? Wtk : Wtv);
    float s = (m == 0) ? 0.125f : 1.0f;
    Wt[(size_t)o * DIN + i] = (_Float16)(W[(size_t)i * DHEAD + o] * s);
}

// ---------------------------------------------------------------------------
// Phase 1: QKV projection GEMM via WMMA.
// One wave computes a 16x64 output tile of one of Q/K/V.
// X:[16384,768] f32 -> Qf,Kf:[16384,64] f16, V transposed Vt:[B][64][4096] f16
// ---------------------------------------------------------------------------
__global__ __launch_bounds__(256) void qkv_gemm(const float* __restrict__ X,
                                                const _Float16* __restrict__ Wtq,
                                                const _Float16* __restrict__ Wtk,
                                                const _Float16* __restrict__ Wtv,
                                                _Float16* __restrict__ Qf,
                                                _Float16* __restrict__ Kf,
                                                _Float16* __restrict__ Vt) {
    const int lane  = threadIdx.x & 31;
    const int laneN = lane & 15;
    const int hi    = lane >> 4;
    const int kbL   = hi << 3;            // C row base (0 or 8)
    const int kB16  = hi << 4;            // B-operand K base (0 or 16)
    const int waveId = blockIdx.x * 8 + (threadIdx.x >> 5);
    const int m    = waveId >> 10;        // 0=Q 1=K 2=V
    const int rt   = waveId & 1023;       // 16-row tile index over 16384 rows
    const int row0 = rt << 4;

    const _Float16* W = (m == 0) ? Wtq : ((m == 1) ? Wtk : Wtv);
    const float* xrow = X + (size_t)(row0 + laneN) * DIN;
    const _Float16* w0 = W + (size_t)( 0 + laneN) * DIN;
    const _Float16* w1 = W + (size_t)(16 + laneN) * DIN;
    const _Float16* w2 = W + (size_t)(32 + laneN) * DIN;
    const _Float16* w3 = W + (size_t)(48 + laneN) * DIN;

    v8f acc0 = {}, acc1 = {}, acc2 = {}, acc3 = {};
    for (int k0 = 0; k0 < DIN; k0 += 32) {
        // A tile: 16x32 slice of X, converted f32 -> f16 in registers
        v8f x0 = *(const v8f*)(xrow + k0 + kB16);
        v8f x1 = *(const v8f*)(xrow + k0 + kB16 + 8);
        V16U a;
#pragma unroll
        for (int i = 0; i < 8; ++i) {
            a.h[0][i] = (_Float16)x0[i];
            a.h[1][i] = (_Float16)x1[i];
        }
        acc0 = wmma16(a.v, ld16(w0, k0 + kB16), acc0);
        acc1 = wmma16(a.v, ld16(w1, k0 + kB16), acc1);
        acc2 = wmma16(a.v, ld16(w2, k0 + kB16), acc2);
        acc3 = wmma16(a.v, ld16(w3, k0 + kB16), acc3);
    }

    // C layout: lane holds (row = kbL+v, col = ntile*16+laneN)
#pragma unroll
    for (int v = 0; v < 8; ++v) {
        int token = row0 + kbL + v;
        if (m == 0) {
            _Float16* q = Qf + (size_t)token * DHEAD + laneN;
            q[0]  = (_Float16)acc0[v];
            q[16] = (_Float16)acc1[v];
            q[32] = (_Float16)acc2[v];
            q[48] = (_Float16)acc3[v];
        } else if (m == 1) {
            _Float16* k = Kf + (size_t)token * DHEAD + laneN;
            k[0]  = (_Float16)acc0[v];
            k[16] = (_Float16)acc1[v];
            k[32] = (_Float16)acc2[v];
            k[48] = (_Float16)acc3[v];
        } else {
            int bb = token >> 12, t = token & (SEQ - 1);
            _Float16* vt = Vt + ((size_t)bb * DHEAD + laneN) * SEQ + t;
            vt[(size_t) 0 * SEQ] = (_Float16)acc0[v];
            vt[(size_t)16 * SEQ] = (_Float16)acc1[v];
            vt[(size_t)32 * SEQ] = (_Float16)acc2[v];
            vt[(size_t)48 * SEQ] = (_Float16)acc3[v];
        }
    }
}

// ---------------------------------------------------------------------------
// Phase 2: register-resident causal flash attention.
// One wave per 16-query tile. Computes S^T = K*Q^T so softmax is lane-local
// (plus one shfl_xor(16) to the partner lane). P^T is rebuilt into the
// B-operand layout with an 8-dword partner exchange. No LDS tiles.
// ---------------------------------------------------------------------------
__global__ __launch_bounds__(256) void flash_attn(const _Float16* __restrict__ Qf,
                                                  const _Float16* __restrict__ Kf,
                                                  const _Float16* __restrict__ Vt,
                                                  float* __restrict__ out) {
    const int lane  = threadIdx.x & 31;
    const int laneN = lane & 15;
    const int hi    = lane >> 4;
    const int kbL   = hi << 3;            // A/C K-run base
    const int kB16  = hi << 4;            // V^T A-operand K-run base
    const int waveId = blockIdx.x * 8 + (threadIdx.x >> 5);
    const int b  = waveId >> 8;           // 256 query tiles per batch
    const int q0 = (waveId & 255) << 4;

    // Q^T as B operand: lane = query column q0+laneN, dims in two 8-runs at kbL, kbL+16
    const _Float16* qrow = Qf + (size_t)((b << 12) + q0 + laneN) * DHEAD;
    v16h qb0 = ldpair(qrow,      kbL, kbL + 16);     // dims  0..31
    v16h qb1 = ldpair(qrow, 32 + kbL, 48 + kbL);     // dims 32..63

    const _Float16* vt0 = Vt + ((size_t)b * DHEAD +  0 + laneN) * SEQ;
    const _Float16* vt1 = Vt + ((size_t)b * DHEAD + 16 + laneN) * SEQ;
    const _Float16* vt2 = Vt + ((size_t)b * DHEAD + 32 + laneN) * SEQ;
    const _Float16* vt3 = Vt + ((size_t)b * DHEAD + 48 + laneN) * SEQ;

    v8f o0 = {}, o1 = {}, o2 = {}, o3 = {};   // O^T: 4 dim-tiles x (16x16)
    float M = -1e30f, S = 0.0f;
    const int qAbs = q0 + laneN;
    const int nkb  = ((q0 + 15) >> 5) + 1;    // causal: only blocks <= diag

    for (int kb = 0; kb < nkb; ++kb) {
        const int kb_abs = kb << 5;

        // S^T tiles: A = K rows (16 keys x 64 dims), B = Q^T
        const _Float16* krow0 = Kf + (size_t)((b << 12) + kb_abs + laneN) * DHEAD;
        const _Float16* krow1 = krow0 + 16 * DHEAD;
        v8f st0 = {}, st1 = {};
        st0 = wmma16(ldpair(krow0,      kbL,      kbL + 16), qb0, st0);
        st0 = wmma16(ldpair(krow0, 32 + kbL, 48 + kbL),      qb1, st0);
        st1 = wmma16(ldpair(krow1,      kbL,      kbL + 16), qb0, st1);
        st1 = wmma16(ldpair(krow1, 32 + kbL, 48 + kbL),      qb1, st1);

        // Causal mask + block max (lane-local; partner lane via shfl_xor 16)
        float s0[8], s1[8];
        float pm = -1e30f;
#pragma unroll
        for (int v = 0; v < 8; ++v) {
            int k0a = kb_abs +      kbL + v;
            int k1a = kb_abs + 16 + kbL + v;
            s0[v] = (k0a <= qAbs) ? st0[v] : -1e30f;
            s1[v] = (k1a <= qAbs) ? st1[v] : -1e30f;
            pm = fmaxf(pm, fmaxf(s0[v], s1[v]));
        }
        pm = fmaxf(pm, __shfl_xor(pm, 16, 32));
        const float newM   = fmaxf(M, pm);
        const float scaleF = __expf(M - newM);

        // P^T = exp(S^T - newM), packed to f16 pairs.
        // Own lane holds keys {kbL+v} of subtile0 (h[0]) and subtile1 (h[1]).
        V16U pown;
        float rs = 0.0f;
#pragma unroll
        for (int v = 0; v < 8; ++v) {
            float p0 = __expf(s0[v] - newM);
            float p1 = __expf(s1[v] - newM);
            pown.h[0][v] = (_Float16)p0;
            pown.h[1][v] = (_Float16)p1;
            rs += p0 + p1;
        }
        rs += __shfl_xor(rs, 16, 32);
        S = S * scaleF + rs;
        M = newM;

        // Rebuild into B-operand layout: lane half hi needs ALL 16 keys of
        // subtile hi. Missing 8 keys live in partner lane (lane^16).
        //   lanes 0-15 : [own sub0 keys 0-7 | partner sub0 keys 8-15]
        //   lanes 16-31: [partner sub1 keys 16-23 | own sub1 keys 24-31]
        V16U pb;
#pragma unroll
        for (int j = 0; j < 4; ++j) {
            int x0 = __shfl_xor(pown.i[j],     16, 32);  // partner subtile0
            int x1 = __shfl_xor(pown.i[4 + j], 16, 32);  // partner subtile1
            pb.i[j]     = hi ? x1 : pown.i[j];
            pb.i[4 + j] = hi ? pown.i[4 + j] : x0;
        }

#pragma unroll
        for (int v = 0; v < 8; ++v) {
            o0[v] *= scaleF; o1[v] *= scaleF;
            o2[v] *= scaleF; o3[v] *= scaleF;
        }

        // O^T += V^T * P^T  (A = V^T dim-tile, 16 dims x 32 keys)
        o0 = wmma16(ld16(vt0, kb_abs + kB16), pb.v, o0);
        o1 = wmma16(ld16(vt1, kb_abs + kB16), pb.v, o1);
        o2 = wmma16(ld16(vt2, kb_abs + kB16), pb.v, o2);
        o3 = wmma16(ld16(vt3, kb_abs + kB16), pb.v, o3);
    }

    // Normalize and store: O^T C-layout -> out[b][q][dim] f32
    const float inv = 1.0f / S;
    float* orow = out + (size_t)((b << 12) + q0 + laneN) * DHEAD;
#pragma unroll
    for (int v = 0; v < 8; ++v) {
        orow[ 0 + kbL + v] = o0[v] * inv;
        orow[16 + kbL + v] = o1[v] * inv;
        orow[32 + kbL + v] = o2[v] * inv;
        orow[48 + kbL + v] = o3[v] * inv;
    }
}

// ---------------------------------------------------------------------------
extern "C" void kernel_launch(void* const* d_in, const int* in_sizes, int n_in,
                              void* d_out, int out_size, void* d_ws, size_t ws_size,
                              hipStream_t stream) {
    (void)in_sizes; (void)n_in; (void)out_size; (void)ws_size;
    const float* X  = (const float*)d_in[0];
    const float* Wq = (const float*)d_in[1];
    const float* Wk = (const float*)d_in[2];
    const float* Wv = (const float*)d_in[3];
    float* out = (float*)d_out;

    // Workspace carve-up (f16 elements): 3 transposed weights, Q, K, V^T
    _Float16* ws  = (_Float16*)d_ws;
    _Float16* Wtq = ws;
    _Float16* Wtk = Wtq + (size_t)DIN * DHEAD;
    _Float16* Wtv = Wtk + (size_t)DIN * DHEAD;
    _Float16* Qf  = Wtv + (size_t)DIN * DHEAD;
    _Float16* Kf  = Qf  + (size_t)BATCH * SEQ * DHEAD;
    _Float16* Vt  = Kf  + (size_t)BATCH * SEQ * DHEAD;

    prep_w<<<(3 * DIN * DHEAD + 255) / 256, 256, 0, stream>>>(Wq, Wk, Wv, Wtq, Wtk, Wtv);
    // 3 matrices * 1024 row-tiles = 3072 waves, 8 waves / block
    qkv_gemm<<<384, 256, 0, stream>>>(X, Wtq, Wtk, Wtv, Qf, Kf, Vt);
    // 4 batches * 256 query tiles = 1024 waves, 8 waves / block
    flash_attn<<<128, 256, 0, stream>>>(Qf, Kf, Vt, out);
}
